// FastRCNN_19937238188159
// MI455X (gfx1250) — matrix operand order repaired and checked by reading, not compiled, not run on hardware
//
#include <hip/hip_runtime.h>
#include <hip/hip_bf16.h>

typedef __attribute__((ext_vector_type(16))) __bf16 v16bf;
typedef __attribute__((ext_vector_type(8)))  float  v8f;

#define NEGV (-1.0e30f)

// ---------- helpers ----------
// round-to-nearest (half-up) f32 -> bf16: 1 add + shift; differs from RNE only
// on exact half-ulp ties, far below bf16 noise for this workload.
__device__ __forceinline__ unsigned f2bf1(float f) {
  union { float f; unsigned u; } c; c.f = f;
  return (c.u + 0x8000u) >> 16;
}
__device__ __forceinline__ unsigned f2bf_pk(float lo, float hi) {
  union { float f; unsigned u; } a, b; a.f = lo; b.f = hi;
  return ((a.u + 0x8000u) >> 16) | ((b.u + 0x8000u) & 0xffff0000u);
}

union FragBF { v16bf v; unsigned u[8]; };

// A fragment (bf16 in memory): pure register reshuffle of two b128 loads.
// rowkb already includes row base + kb + (lane>=16 ? 8 : 0).
__device__ __forceinline__ v16bf a_frag(const unsigned short* rowkb) {
  const uint4* p = (const uint4*)rowkb;          // 16B-aligned at all call sites
  uint4 q0 = p[0];                               // K = base .. base+7
  uint4 q1 = p[2];                               // K = base+16 .. base+23
  FragBF f;
  f.u[0] = q0.x; f.u[1] = q0.y; f.u[2] = q0.z; f.u[3] = q0.w;
  f.u[4] = q1.x; f.u[5] = q1.y; f.u[6] = q1.z; f.u[7] = q1.w;
  return f.v;
}

// Raw fp32 weight quad (two 32B runs: K+0..7 and K+16..23) -- loaded early (prefetch), converted late.
struct BRaw { float4 r0, r1, r2, r3; };
__device__ __forceinline__ BRaw b_raw(const float* rowkb) {
  const float4* p = (const float4*)rowkb;        // 16B-aligned at all call sites
  BRaw b; b.r0 = p[0]; b.r1 = p[1]; b.r2 = p[4]; b.r3 = p[5];
  return b;
}
__device__ __forceinline__ v16bf b_cvt(const BRaw& b) {
  FragBF f;
  f.u[0] = f2bf_pk(b.r0.x, b.r0.y); f.u[1] = f2bf_pk(b.r0.z, b.r0.w);
  f.u[2] = f2bf_pk(b.r1.x, b.r1.y); f.u[3] = f2bf_pk(b.r1.z, b.r1.w);
  f.u[4] = f2bf_pk(b.r2.x, b.r2.y); f.u[5] = f2bf_pk(b.r2.z, b.r2.w);
  f.u[6] = f2bf_pk(b.r3.x, b.r3.y); f.u[7] = f2bf_pk(b.r3.z, b.r3.w);
  return f.v;
}

template<bool RELU, bool OUTBF16>
__device__ __forceinline__ void store_tile(void* out, const float* bias, v8f acc,
                                           int M, int N, int mtile, int ntile, int lane) {
  int n = ntile * 16 + (lane & 15);
  float bv = bias[min(n, N - 1)];
  int m0 = mtile * 16 + ((lane & 16) ? 8 : 0);
#pragma unroll
  for (int r = 0; r < 8; ++r) {
    int m = m0 + r;
    if (m < M && n < N) {
      float v = acc[r] + bv;
      if (RELU) v = fmaxf(v, 0.0f);
      unsigned off = (unsigned)m * (unsigned)N + (unsigned)n;   // fits 32-bit
      if (OUTBF16) ((unsigned short*)out)[off] = (unsigned short)f2bf1(v);
      else         ((float*)out)[off] = v;
    }
  }
}

// ---------- GEMM: one wave per 16x16 tile, software-pipelined K loop ----------
template<bool RELU, bool OUTBF16>
__global__ void gemm_bf16_t1(const unsigned short* A, const float* W, const float* bias,
                             void* out, int M, int N, int K) {
  int lane = threadIdx.x & 31;
  int wave = blockIdx.x * (blockDim.x >> 5) + (threadIdx.x >> 5);
  int mtiles = (M + 15) >> 4, ntiles = (N + 15) >> 4;
  if (wave >= mtiles * ntiles) return;
  int mt = wave % mtiles, nt = wave / mtiles;
  // clamp row/col: OOB rows/cols only feed D elements the store guard discards
  const unsigned short* arow =
      A + (size_t)min(mt * 16 + (lane & 15), M - 1) * (size_t)K + ((lane & 16) ? 8 : 0);
  const float* wrow =
      W + (size_t)min(nt * 16 + (lane & 15), N - 1) * (size_t)K + ((lane & 16) ? 8 : 0);
  v8f acc;
#pragma unroll
  for (int i = 0; i < 8; ++i) acc[i] = 0.0f;
  v16bf a = a_frag(arow);
  BRaw  br = b_raw(wrow);
  for (int kb = 32; kb < K; kb += 32) {
    v16bf an = a_frag(arow + kb);     // next-step loads issue before consume
    BRaw  bn = b_raw(wrow + kb);
    v16bf b = b_cvt(br);
    acc = __builtin_amdgcn_wmma_f32_16x16x32_bf16(false, a, false, b, (short)0, acc, false, false);
    a = an; br = bn;
  }
  {
    v16bf b = b_cvt(br);
    acc = __builtin_amdgcn_wmma_f32_16x16x32_bf16(false, a, false, b, (short)0, acc, false, false);
  }
  store_tile<RELU, OUTBF16>(out, bias, acc, M, N, mt, nt, lane);
}

// ---------- GEMM, M=300: wave owns one N-tile, loops all 19 M-tiles so the
// HBM weight stream is read exactly once (fc6 = 616MB fp32 dominates roofline;
// the 22.6MB bf16 activations are L2-resident).  Weights are double-buffered
// raw + prefetched ~1KB ahead; A fragments are double-buffered across t. ----------
#define MT300 19
template<bool RELU, bool OUTBF16>
__global__ void gemm_bf16_m300(const unsigned short* A, const float* W, const float* bias,
                               void* out, int M, int N, int K) {
  int lane = threadIdx.x & 31;
  int nt = blockIdx.x * (blockDim.x >> 5) + (threadIdx.x >> 5);
  int ntiles = (N + 15) >> 4;
  if (nt >= ntiles) return;
  const unsigned short* abase = A + ((lane & 16) ? 8 : 0);
  unsigned aoff[MT300];
#pragma unroll
  for (int t = 0; t < MT300; ++t)
    aoff[t] = (unsigned)min(t * 16 + (lane & 15), M - 1) * (unsigned)K;
  const float* wrow =
      W + (size_t)min(nt * 16 + (lane & 15), N - 1) * (size_t)K + ((lane & 16) ? 8 : 0);
  v8f acc[MT300];
#pragma unroll
  for (int t = 0; t < MT300; ++t)
#pragma unroll
    for (int i = 0; i < 8; ++i) acc[t][i] = 0.0f;
  BRaw br = b_raw(wrow);
  for (int kb = 0; kb < K; kb += 32) {
    int nkb = (kb + 32 < K) ? (kb + 32) : kb;          // branchless tail
    __builtin_prefetch(wrow + kb + 256, 0, 0);          // global_prefetch_b8, ~1KB ahead
    BRaw bn = b_raw(wrow + nkb);
    v16bf b = b_cvt(br);
    v16bf a = a_frag(abase + aoff[0] + kb);
#pragma unroll
    for (int t = 0; t < MT300; ++t) {
      v16bf an = (t + 1 < MT300) ? a_frag(abase + aoff[t + 1] + kb) : a;
      acc[t] = __builtin_amdgcn_wmma_f32_16x16x32_bf16(false, a, false, b, (short)0, acc[t], false, false);
      a = an;
    }
    br = bn;
  }
  for (int t = 0; t < MT300; ++t)
    store_tile<RELU, OUTBF16>(out, bias, acc[t], M, N, t, nt, lane);
}

// ---------- im2col for 3x3 SAME conv (768ch, 38x50) -> bf16 [1900][6912] ----------
__global__ void im2col_kernel(const float* feat, unsigned short* A0) {
  int i = blockIdx.x * blockDim.x + threadIdx.x;
  if (i >= 1900 * 6912) return;
  int m = i / 6912, k = i - m * 6912;
  int h = m / 50, w = m - h * 50;
  int c = k / 9, r = k - c * 9;
  int kh = r / 3, kw = r - kh * 3;
  int ih = h + kh - 1, iw = w + kw - 1;
  float v = 0.0f;
  if (ih >= 0 && ih < 38 && iw >= 0 && iw < 50) v = feat[(c * 38 + ih) * 50 + iw];
  A0[i] = (unsigned short)f2bf1(v);
}

// ---------- anchors + bbox_transform_inv + clip + min-size + fg softmax ----------
__global__ void proposals_kernel(const float* scorebuf /*[1900][24]*/,
                                 const float* deltabuf /*[1900][48]*/,
                                 const float* im_info,
                                 float* proposals, float* pscore) {
  int i = blockIdx.x * blockDim.x + threadIdx.x;
  if (i >= 22800) return;
  int m = i / 12, a = i - m * 12;
  int h = m / 50, w = m - h * 50;
  int ri = a >> 2, si = a & 3;
  const float ratios[3] = {0.5f, 1.0f, 2.0f};
  const float scales[4] = {4.0f, 8.0f, 16.0f, 32.0f};
  float rr = ratios[ri], ss = scales[si];
  float ws_ = roundf(sqrtf(256.0f / rr));
  float hs_ = roundf(ws_ * rr);
  float Wa = ws_ * ss, Ha = hs_ * ss;
  float cx = 7.5f + w * 16.0f, cy = 7.5f + h * 16.0f;
  float ax1 = cx - 0.5f * (Wa - 1.f), ay1 = cy - 0.5f * (Ha - 1.f);
  float ax2 = cx + 0.5f * (Wa - 1.f), ay2 = cy + 0.5f * (Ha - 1.f);
  float bw = ax2 - ax1 + 1.f, bh = ay2 - ay1 + 1.f;
  float bcx = ax1 + 0.5f * bw, bcy = ay1 + 0.5f * bh;
  const float* d = deltabuf + (size_t)m * 48 + a * 4;
  float pcx = d[0] * bw + bcx, pcy = d[1] * bh + bcy;
  float pw = expf(d[2]) * bw, ph = expf(d[3]) * bh;
  float x1 = pcx - 0.5f * pw, y1 = pcy - 0.5f * ph;
  float x2 = pcx + 0.5f * pw, y2 = pcy + 0.5f * ph;
  float im_h = im_info[0], im_w = im_info[1], sc = im_info[2];
  x1 = fminf(fmaxf(x1, 0.f), im_w - 1.f);
  x2 = fminf(fmaxf(x2, 0.f), im_w - 1.f);
  y1 = fminf(fmaxf(y1, 0.f), im_h - 1.f);
  y2 = fminf(fmaxf(y2, 0.f), im_h - 1.f);
  float fg = scorebuf[(size_t)m * 24 + 12 + a];
  float bg = scorebuf[(size_t)m * 24 + a];
  float p = 1.0f / (1.0f + expf(bg - fg));  // softmax over {bg,fg}
  float ms = 16.0f * sc;
  if (!(((x2 - x1 + 1.f) >= ms) && ((y2 - y1 + 1.f) >= ms))) p = NEGV;
  proposals[(size_t)i * 4 + 0] = x1;
  proposals[(size_t)i * 4 + 1] = y1;
  proposals[(size_t)i * 4 + 2] = x2;
  proposals[(size_t)i * 4 + 3] = y2;
  pscore[i] = p;
}

// ---------- global bitonic sort (descending), 32768 padded ----------
__global__ void sort_init(const float* pscore, float* key, int* idx) {
  int i = blockIdx.x * blockDim.x + threadIdx.x;
  if (i >= 32768) return;
  key[i] = (i < 22800) ? pscore[i] : -3.4e38f;
  idx[i] = i;
}
__global__ void sort_pass(float* key, int* idx, int ksz, int j) {
  int i = blockIdx.x * blockDim.x + threadIdx.x;
  if (i >= 32768) return;
  int l = i ^ j;
  if (l > i) {
    bool up = ((i & ksz) == 0);
    float ki = key[i], kl = key[l];
    bool sw = up ? (ki < kl) : (ki > kl);  // descending overall
    if (sw) {
      key[i] = kl; key[l] = ki;
      int t = idx[i]; idx[i] = idx[l]; idx[l] = t;
    }
  }
}
__global__ void gather_top(const float* key, const int* idx, const float* proposals,
                           float* top_b, float* top_s) {
  int i = blockIdx.x * blockDim.x + threadIdx.x;
  if (i >= 6000) return;
  top_s[i] = key[i];
  int id = idx[i];
#pragma unroll
  for (int c = 0; c < 4; ++c) top_b[(size_t)i * 4 + c] = proposals[(size_t)id * 4 + c];
}

// ---------- NMS: single workgroup, 300 serial argmax+suppress iterations ----------
__global__ void nms_kernel(const float* top_b, const float* top_s,
                           float* rois_out, float* rois_ws, int* valid_ws) {
  __shared__ float sc[6000];
  __shared__ float rv[256];
  __shared__ int   ri[256];
  __shared__ float bb[4];
  __shared__ int   bok;
  int tid = threadIdx.x;
  for (int t = tid; t < 6000; t += 256) sc[t] = top_s[t];
  __syncthreads();
  for (int it = 0; it < 300; ++it) {
    float best = -3.4e38f; int bidx = 0;
    for (int t = tid; t < 6000; t += 256) {
      float s = sc[t];
      if (s > best) { best = s; bidx = t; }
    }
    rv[tid] = best; ri[tid] = bidx;
    __syncthreads();
    for (int s = 128; s > 0; s >>= 1) {
      if (tid < s) {
        if (rv[tid + s] > rv[tid] || (rv[tid + s] == rv[tid] && ri[tid + s] < ri[tid])) {
          rv[tid] = rv[tid + s]; ri[tid] = ri[tid + s];
        }
      }
      __syncthreads();
    }
    if (tid == 0) {
      int j = ri[0];
      int ok = (rv[0] > 0.5f * NEGV) ? 1 : 0;
      bok = ok;
      bb[0] = top_b[(size_t)j * 4 + 0]; bb[1] = top_b[(size_t)j * 4 + 1];
      bb[2] = top_b[(size_t)j * 4 + 2]; bb[3] = top_b[(size_t)j * 4 + 3];
      sc[j] = NEGV;
      valid_ws[it] = ok;
      float o1 = ok ? bb[0] : 0.f, o2 = ok ? bb[1] : 0.f;
      float o3 = ok ? bb[2] : 0.f, o4 = ok ? bb[3] : 0.f;
      rois_out[it * 5 + 0] = 0.f; rois_out[it * 5 + 1] = o1; rois_out[it * 5 + 2] = o2;
      rois_out[it * 5 + 3] = o3;  rois_out[it * 5 + 4] = o4;
      rois_ws[it * 5 + 0] = 0.f;  rois_ws[it * 5 + 1] = o1;  rois_ws[it * 5 + 2] = o2;
      rois_ws[it * 5 + 3] = o3;   rois_ws[it * 5 + 4] = o4;
    }
    __syncthreads();
    if (bok) {
      float bx1 = bb[0], by1 = bb[1], bx2 = bb[2], by2 = bb[3];
      float barea = (bx2 - bx1 + 1.f) * (by2 - by1 + 1.f);
      for (int t = tid; t < 6000; t += 256) {
        float x1 = top_b[(size_t)t * 4 + 0], y1 = top_b[(size_t)t * 4 + 1];
        float x2 = top_b[(size_t)t * 4 + 2], y2 = top_b[(size_t)t * 4 + 3];
        float iw = fmaxf(fminf(x2, bx2) - fmaxf(x1, bx1) + 1.f, 0.f);
        float ih = fmaxf(fminf(y2, by2) - fmaxf(y1, by1) + 1.f, 0.f);
        float inter = iw * ih;
        float area = (x2 - x1 + 1.f) * (y2 - y1 + 1.f);
        float iou = inter / (area + barea - inter);
        if (iou > 0.7f) sc[t] = NEGV;
      }
    }
    __syncthreads();
  }
}

// ---------- ROI max-pool 7x7, writes bf16 (WMMA A-operand of fc6) ----------
__global__ void roi_pool_kernel(const float* feat, const float* rois_ws, const int* valid_ws,
                                unsigned short* pooled) {
  int r = blockIdx.x;
  int v = valid_ws[r];
  float x1 = roundf(rois_ws[r * 5 + 1] * 0.0625f);
  float y1 = roundf(rois_ws[r * 5 + 2] * 0.0625f);
  float x2 = roundf(rois_ws[r * 5 + 3] * 0.0625f);
  float y2 = roundf(rois_ws[r * 5 + 4] * 0.0625f);
  float rw = fmaxf(x2 - x1 + 1.f, 1.f);
  float rh = fmaxf(y2 - y1 + 1.f, 1.f);
  float bw = rw / 7.f, bh = rh / 7.f;
  for (int e = threadIdx.x; e < 37632; e += blockDim.x) {
    int c = e / 49, bin = e - c * 49;
    int ph = bin / 7, pw = bin - ph * 7;
    int hs  = (int)fminf(fmaxf(floorf(ph * bh) + y1, 0.f), 38.f);
    int he  = (int)fminf(fmaxf(ceilf((ph + 1) * bh) + y1, 0.f), 38.f);
    int wsb = (int)fminf(fmaxf(floorf(pw * bw) + x1, 0.f), 50.f);
    int web = (int)fminf(fmaxf(ceilf((pw + 1) * bw) + x1, 0.f), 50.f);
    float mval = NEGV;
    const float* fc_ = feat + (size_t)c * 38 * 50;
    for (int hh = hs; hh < he; ++hh)
      for (int ww = wsb; ww < web; ++ww)
        mval = fmaxf(mval, fc_[hh * 50 + ww]);
    if (mval <= 0.5f * NEGV) mval = 0.f;
    if (!v) mval = 0.f;
    pooled[(size_t)r * 37632 + e] = (unsigned short)f2bf1(mval);
  }
}

// ---------- softmax over 21 classes, one wave per row ----------
__global__ void softmax21_kernel(const float* logits, float* out) {
  int row = blockIdx.x, lane = threadIdx.x;
  float v = (lane < 21) ? logits[(size_t)row * 21 + lane] : -3.4e38f;
  float mx = v;
  for (int o = 16; o > 0; o >>= 1) mx = fmaxf(mx, __shfl_xor(mx, o, 32));
  float e = (lane < 21) ? expf(v - mx) : 0.f;
  float s = e;
  for (int o = 16; o > 0; o >>= 1) s += __shfl_xor(s, o, 32);
  if (lane < 21) out[(size_t)row * 21 + lane] = e / s;
}

// =======================================================================
extern "C" void kernel_launch(void* const* d_in, const int* in_sizes, int n_in,
                              void* d_out, int out_size, void* d_ws, size_t ws_size,
                              hipStream_t stream) {
  (void)in_sizes; (void)n_in; (void)out_size; (void)ws_size;
  const float* features = (const float*)d_in[0];
  const float* im_info  = (const float*)d_in[1];
  const float* conv1_w  = (const float*)d_in[2];
  const float* conv1_b  = (const float*)d_in[3];
  const float* score_w  = (const float*)d_in[4];
  const float* score_b  = (const float*)d_in[5];
  const float* bbox_w   = (const float*)d_in[6];
  const float* bbox_b   = (const float*)d_in[7];
  const float* fc6_w    = (const float*)d_in[8];
  const float* fc6_b    = (const float*)d_in[9];
  const float* fc7_w    = (const float*)d_in[10];
  const float* fc7_b    = (const float*)d_in[11];
  const float* cls_w    = (const float*)d_in[12];
  const float* cls_b    = (const float*)d_in[13];
  const float* breg_w   = (const float*)d_in[14];
  const float* breg_b   = (const float*)d_in[15];

  float* out      = (float*)d_out;
  float* cls_prob = out;            // 300*21
  float* bbox_out = out + 6300;     // 300*84
  float* rois_out = out + 31500;    // 300*5

  char* ws = (char*)d_ws;
  size_t off = 0;
  auto take = [&](size_t bytes) { void* p = ws + off; off = (off + bytes + 255) & ~(size_t)255; return p; };
  unsigned short* A0     = (unsigned short*)take((size_t)1900 * 6912 * 2);
  unsigned short* rpn1   = (unsigned short*)take((size_t)1900 * 512 * 2);
  float* scores    = (float*)take((size_t)1900 * 24 * 4);
  float* deltas    = (float*)take((size_t)1900 * 48 * 4);
  float* proposals = (float*)take((size_t)22800 * 4 * 4);
  float* pscore    = (float*)take((size_t)22800 * 4);
  float* skey      = (float*)take((size_t)32768 * 4);
  int*   sidx      = (int*)take((size_t)32768 * 4);
  float* top_b     = (float*)take((size_t)6000 * 4 * 4);
  float* top_s     = (float*)take((size_t)6000 * 4);
  float* rois_ws   = (float*)take((size_t)300 * 5 * 4);
  int*   valid_ws  = (int*)take((size_t)300 * 4);
  unsigned short* pooled = (unsigned short*)take((size_t)300 * 37632 * 2);
  unsigned short* x1b    = (unsigned short*)take((size_t)300 * 4096 * 2);
  unsigned short* x2b    = (unsigned short*)take((size_t)300 * 4096 * 2);
  float* logits    = (float*)take((size_t)300 * 21 * 4);

  // 1) im2col (3x3 SAME) -> bf16
  im2col_kernel<<<(1900 * 6912 + 255) / 256, 256, 0, stream>>>(features, A0);

  // 2) conv1 as GEMM: M=1900, K=6912, N=512, ReLU, bf16 out  (119*32 waves)
  gemm_bf16_t1<true, true><<<(119 * 32 + 7) / 8, 256, 0, stream>>>(A0, conv1_w, conv1_b, rpn1,
                                                                   1900, 512, 6912);
  // 3) 1x1 score conv: N=24 (2 tiles)
  gemm_bf16_t1<false, false><<<(119 * 2 + 7) / 8, 256, 0, stream>>>(rpn1, score_w, score_b, scores,
                                                                    1900, 24, 512);
  // 4) 1x1 bbox conv: N=48
  gemm_bf16_t1<false, false><<<(119 * 3 + 7) / 8, 256, 0, stream>>>(rpn1, bbox_w, bbox_b, deltas,
                                                                    1900, 48, 512);
  // 5) proposals
  proposals_kernel<<<(22800 + 255) / 256, 256, 0, stream>>>(scores, deltas, im_info,
                                                            proposals, pscore);
  // 6) exact top-6000 via global bitonic sort (grid sync = per-pass launches)
  sort_init<<<128, 256, 0, stream>>>(pscore, skey, sidx);
  for (int k = 2; k <= 32768; k <<= 1)
    for (int j = k >> 1; j > 0; j >>= 1)
      sort_pass<<<128, 256, 0, stream>>>(skey, sidx, k, j);
  gather_top<<<(6000 + 255) / 256, 256, 0, stream>>>(skey, sidx, proposals, top_b, top_s);

  // 7) NMS -> rois (output 3) + ws copies
  nms_kernel<<<1, 256, 0, stream>>>(top_b, top_s, rois_out, rois_ws, valid_ws);

  // 8) ROI max-pool -> bf16 pooled [300][37632]
  roi_pool_kernel<<<300, 256, 0, stream>>>(features, rois_ws, valid_ws, pooled);

  // 9) fc6: M=300, K=37632, N=4096 (weights 616MB stream once), ReLU, bf16 out
  gemm_bf16_m300<true, true><<<(256 + 7) / 8, 256, 0, stream>>>(pooled, fc6_w, fc6_b, x1b,
                                                                300, 4096, 37632);
  // 10) fc7: M=300, K=4096, N=4096, ReLU, bf16 out
  gemm_bf16_m300<true, true><<<(256 + 7) / 8, 256, 0, stream>>>(x1b, fc7_w, fc7_b, x2b,
                                                                300, 4096, 4096);
  // 11) cls logits: N=21
  gemm_bf16_m300<false, false><<<1, 256, 0, stream>>>(x2b, cls_w, cls_b, logits, 300, 21, 4096);
  // 12) breg -> output 2 directly: N=84
  gemm_bf16_m300<false, false><<<1, 256, 0, stream>>>(x2b, breg_w, breg_b, bbox_out, 300, 84, 4096);
  // 13) softmax -> output 1
  softmax21_kernel<<<300, 32, 0, stream>>>(logits, cls_prob);
}
